// ViT_7086696038478
// MI455X (gfx1250) — matrix operand order repaired and checked by reading, not compiled, not run on hardware
//
#include <hip/hip_runtime.h>
#include <hip/hip_bf16.h>

// ---------------------------------------------------------------------------
// Problem constants (from the reference)
// ---------------------------------------------------------------------------
#define BSZ   512
#define P196  196
#define NPOS  197          // P + 1
#define PD    768
#define LD    150
#define EDIM  300
#define HHEADS 2
#define CDIM  1000
#define TTOK  (BSZ * NPOS) // 100864
#define MPROJ (BSZ * P196) // 100352
#define ATTN_SCALE 0.05773502691896258f   // 1/sqrt(300)

typedef __attribute__((ext_vector_type(16))) _Float16 v16h;
typedef __attribute__((ext_vector_type(8)))  float    v8f;
typedef __attribute__((ext_vector_type(4)))  _Float16 v4h;

// ---------------------------------------------------------------------------
// Wave32 WMMA GEMM:  C = epilogue(A[MxK](f32) * B[KxN](f32))
//   - f32 -> f16 conversion at LDS staging, f32 accumulate
//   - 256 threads = 8 waves; block tile 64x64; each wave owns two 16x16 tiles
//     sharing one A fragment (2x v_wmma per k-step per wave)
//   - register-staged pipeline: next tile's global loads issue under the WMMA
//   - fast path staging: float4 (A) / float2 (B) loads, no per-element branches
// ---------------------------------------------------------------------------
template<typename TC, bool RELU, bool REMAP>
__global__ __launch_bounds__(256)
void gemm_wmma_kernel(const float* __restrict__ A, int lda,
                      const float* __restrict__ B, int ldb,
                      TC* __restrict__ C, int ldc,
                      const float* __restrict__ bias,
                      int M, int N, int K, int row_group)
{
    constexpr int BM = 64, BN = 64, BK = 32;
    constexpr int LAS = 36;   // LDS k-stride (halves), padded
    constexpr int LBS = 36;

    __shared__ _Float16 Ah[BM * LAS];
    __shared__ _Float16 Bh[BN * LBS];

    const int tid    = threadIdx.x;
    const int wave   = tid >> 5;
    const int lane   = tid & 31;
    const int waveM  = wave & 3;          // 4 tiles along M
    const int waveNb = (wave >> 2) * 32;  // 2 wave-groups along N, 2 tiles each
    const int row16  = lane & 15;
    const int half   = lane >> 4;

    const int blockM = blockIdx.y * BM;
    const int blockN = blockIdx.x * BN;

    // staging roles
    const int amL = tid >> 2;           // 0..63  (A row in tile)
    const int akQ = (tid & 3) * 8;      // 0,8,16,24 (A k-offset, 8 elems)
    const int bkL = tid >> 3;           // 0..31  (B k-row in tile)
    const int bnQ = (tid & 7) * 8;      // 0..56  (B n-offset, 8 elems)

    float ra[8], rb[8];

    auto loadA = [&](int k0) {
        const int  gm   = blockM + amL;
        const long base = (long)gm * lda + k0 + akQ;
        if (gm < M && (k0 + akQ + 7) < K) {
            const float4 p0 = *(const float4*)(A + base);
            const float4 p1 = *(const float4*)(A + base + 4);
            ra[0] = p0.x; ra[1] = p0.y; ra[2] = p0.z; ra[3] = p0.w;
            ra[4] = p1.x; ra[5] = p1.y; ra[6] = p1.z; ra[7] = p1.w;
        } else {
            #pragma unroll
            for (int j = 0; j < 8; ++j) {
                const int k = k0 + akQ + j;
                ra[j] = (gm < M && k < K) ? A[(long)gm * lda + k] : 0.0f;
            }
        }
    };
    auto loadB = [&](int k0) {
        const int k  = k0 + bkL;
        const int gn = blockN + bnQ;
        if (k < K && (gn + 7) < N) {
            const float2* p = (const float2*)(B + (long)k * ldb + gn);
            const float2 q0 = p[0], q1 = p[1], q2 = p[2], q3 = p[3];
            rb[0] = q0.x; rb[1] = q0.y; rb[2] = q1.x; rb[3] = q1.y;
            rb[4] = q2.x; rb[5] = q2.y; rb[6] = q3.x; rb[7] = q3.y;
        } else {
            #pragma unroll
            for (int j = 0; j < 8; ++j)
                rb[j] = (k < K && (gn + j) < N) ? B[(long)k * ldb + gn + j] : 0.0f;
        }
    };
    auto stageStore = [&]() {
        // A: 8 contiguous halves -> two 8-byte packed stores
        v4h h0 = { (_Float16)ra[0], (_Float16)ra[1], (_Float16)ra[2], (_Float16)ra[3] };
        v4h h1 = { (_Float16)ra[4], (_Float16)ra[5], (_Float16)ra[6], (_Float16)ra[7] };
        *(v4h*)(&Ah[amL * LAS + akQ])     = h0;
        *(v4h*)(&Ah[amL * LAS + akQ + 4]) = h1;
        // B: transpose into [n][k] rows
        #pragma unroll
        for (int j = 0; j < 8; ++j)
            Bh[(bnQ + j) * LBS + bkL] = (_Float16)rb[j];
    };

    v8f acc0 = {}, acc1 = {};
    loadA(0);
    loadB(0);

    for (int k0 = 0; k0 < K; k0 += BK) {
        stageStore();
        __syncthreads();

        // fragments in contiguous k order: element s of lane half h holds k = 16h + s (A and B alike)
        v16h afrag, bfrag0, bfrag1;
        const int am = waveM * 16 + row16;
        const int bn = waveNb + row16;
        #pragma unroll
        for (int vv = 0; vv < 8; ++vv) {
            const int ka = half * 16 + vv * 2;
            afrag[2 * vv]      = Ah[am * LAS + ka];
            afrag[2 * vv + 1]  = Ah[am * LAS + ka + 1];
            const int kb = half * 16 + vv * 2;
            bfrag0[2 * vv]     = Bh[bn * LBS + kb];
            bfrag0[2 * vv + 1] = Bh[bn * LBS + kb + 1];
            bfrag1[2 * vv]     = Bh[(bn + 16) * LBS + kb];
            bfrag1[2 * vv + 1] = Bh[(bn + 16) * LBS + kb + 1];
        }

        // issue next tile's global loads under the WMMAs
        if (k0 + BK < K) { loadA(k0 + BK); loadB(k0 + BK); }

        acc0 = __builtin_amdgcn_wmma_f32_16x16x32_f16(
                   false, afrag, false, bfrag0, (short)0, acc0, false, false);
        acc1 = __builtin_amdgcn_wmma_f32_16x16x32_f16(
                   false, afrag, false, bfrag1, (short)0, acc1, false, false);
        __syncthreads();
    }

    // epilogue: C layout VGPR i -> M = i + half*8, N = lane%16 (per 16x16 tile)
    const int gn0 = blockN + waveNb + row16;
    const int gn1 = gn0 + 16;
    #pragma unroll
    for (int i = 0; i < 8; ++i) {
        const int gm = blockM + waveM * 16 + half * 8 + i;
        if (gm < M) {
            long row = gm;
            if (REMAP) row = (long)gm + gm / row_group + 1;
            if (gn0 < N) {
                float v = acc0[i];
                if (bias) v += bias[gn0];
                if (RELU) v = fmaxf(v, 0.0f);
                C[row * (long)ldc + gn0] = (TC)v;
            }
            if (gn1 < N) {
                float v = acc1[i];
                if (bias) v += bias[gn1];
                if (RELU) v = fmaxf(v, 0.0f);
                C[row * (long)ldc + gn1] = (TC)v;
            }
        }
    }
}

// ---------------------------------------------------------------------------
// seq fill: cls row (cls_emb | pos_tab[0]) and positional halves of tokens
// ---------------------------------------------------------------------------
__global__ void fill_seq_kernel(float* __restrict__ seq,
                                const float* __restrict__ pos_tab,
                                const float* __restrict__ cls_emb)
{
    long t = (long)blockIdx.x * 256 + threadIdx.x;
    const long total = (long)BSZ * NPOS * LD;
    if (t >= total) return;
    const int  j = (int)(t % LD);
    const long r = t / LD;
    const int  n = (int)(r % NPOS);
    const long b = r / NPOS;
    float* row = seq + (b * NPOS + n) * (long)EDIM;
    if (n == 0) {
        row[j]      = cls_emb[j];
        row[LD + j] = pos_tab[j];
    } else {
        row[LD + j] = pos_tab[(long)n * LD + j];
    }
}

// ---------------------------------------------------------------------------
// Attention, row 0 only: scores -> softmax -> context, per (b,h)
// ---------------------------------------------------------------------------
__global__ __launch_bounds__(256)
void attn_row0_kernel(const _Float16* __restrict__ Kbuf,
                      const _Float16* __restrict__ Vbuf,
                      const float*    __restrict__ Q0,
                      float*          __restrict__ msa0)
{
    __shared__ float q[EDIM + 4];
    __shared__ float s[NPOS + 3];
    __shared__ float red;

    const int b = blockIdx.x >> 1;
    const int h = blockIdx.x & 1;
    const float*    Qr = Q0   + ((long)h * BSZ + b) * EDIM;
    const _Float16* Kp = Kbuf + ((long)h * TTOK + (long)b * NPOS) * EDIM;
    const _Float16* Vp = Vbuf + ((long)h * TTOK + (long)b * NPOS) * EDIM;

    for (int j = threadIdx.x; j < EDIM; j += 256) q[j] = Qr[j];
    __syncthreads();

    const int m = threadIdx.x;
    if (m < NPOS) {
        float a = 0.0f;
        for (int d = 0; d < EDIM; ++d) a += q[d] * (float)Kp[(long)m * EDIM + d];
        s[m] = a * ATTN_SCALE;
    }
    __syncthreads();
    if (threadIdx.x == 0) {
        float mx = s[0];
        for (int i = 1; i < NPOS; ++i) mx = fmaxf(mx, s[i]);
        red = mx;
    }
    __syncthreads();
    if (m < NPOS) s[m] = __expf(s[m] - red);
    __syncthreads();
    if (threadIdx.x == 0) {
        float sm = 0.0f;
        for (int i = 0; i < NPOS; ++i) sm += s[i];
        red = 1.0f / sm;
    }
    __syncthreads();
    if (m < NPOS) s[m] *= red;
    __syncthreads();

    for (int e = threadIdx.x; e < EDIM; e += 256) {
        float a = 0.0f;
        for (int mm = 0; mm < NPOS; ++mm) a += s[mm] * (float)Vp[(long)mm * EDIM + e];
        msa0[(long)b * (HHEADS * EDIM) + h * EDIM + e] = a;
    }
}

// ---------------------------------------------------------------------------
// BN statistics over a flat [count] array: out[0]=mean, out[1]=rsqrt(var+eps)
// ---------------------------------------------------------------------------
__global__ __launch_bounds__(256)
void bn_stats_kernel(const float* __restrict__ x, int count, float* __restrict__ out)
{
    __shared__ float ssum[256];
    __shared__ float ssq[256];
    float s = 0.0f, q = 0.0f;
    for (int i = threadIdx.x; i < count; i += 256) {
        const float v = x[i];
        s += v; q += v * v;
    }
    ssum[threadIdx.x] = s; ssq[threadIdx.x] = q;
    __syncthreads();
    for (int st = 128; st > 0; st >>= 1) {
        if (threadIdx.x < st) {
            ssum[threadIdx.x] += ssum[threadIdx.x + st];
            ssq[threadIdx.x]  += ssq[threadIdx.x + st];
        }
        __syncthreads();
    }
    if (threadIdx.x == 0) {
        const float mean = ssum[0] / count;
        const float var  = ssq[0] / count - mean * mean;
        out[0] = mean;
        out[1] = rsqrtf(var + 1e-5f);
    }
}

// cat1[b, 0:600] = bn(msa0);  cat1[b, 600:900] = seq[b, n=0, :]
__global__ void build_cat1_kernel(float* __restrict__ cat1,
                                  const float* __restrict__ msa0,
                                  const float* __restrict__ seq,
                                  const float* __restrict__ st,
                                  const float* __restrict__ g0,
                                  const float* __restrict__ beta0)
{
    const int t = blockIdx.x * 256 + threadIdx.x;
    if (t >= BSZ * 900) return;
    const int b = t / 900, j = t % 900;
    float v;
    if (j < 600) v = (msa0[b * 600 + j] - st[0]) * st[1] * g0[0] + beta0[0];
    else         v = seq[(long)b * NPOS * EDIM + (j - 600)];
    cat1[t] = v;
}

// cat2[b, 0:300] = bn(y1);  cat2[b, 300:1200] = cat1[b, :]
__global__ void build_cat2_kernel(float* __restrict__ cat2,
                                  const float* __restrict__ y1,
                                  const float* __restrict__ cat1,
                                  const float* __restrict__ st,
                                  const float* __restrict__ g1,
                                  const float* __restrict__ beta1)
{
    const int t = blockIdx.x * 256 + threadIdx.x;
    if (t >= BSZ * 1200) return;
    const int b = t / 1200, j = t % 1200;
    float v;
    if (j < 300) v = (y1[b * 300 + j] - st[0]) * st[1] * g1[0] + beta1[0];
    else         v = cat1[b * 900 + (j - 300)];
    cat2[t] = v;
}

// ---------------------------------------------------------------------------
static inline int ceil_div(int a, int b) { return (a + b - 1) / b; }

extern "C" void kernel_launch(void* const* d_in, const int* in_sizes, int n_in,
                              void* d_out, int out_size, void* d_ws, size_t ws_size,
                              hipStream_t stream)
{
    (void)in_sizes; (void)n_in; (void)out_size; (void)ws_size;

    const float* x      = (const float*)d_in[0];
    const float* W_proj = (const float*)d_in[1];
    const float* b_proj = (const float*)d_in[2];
    const float* pos_tab= (const float*)d_in[3];
    const float* cls_emb= (const float*)d_in[4];
    const float* Wq     = (const float*)d_in[5];
    const float* Wk     = (const float*)d_in[6];
    const float* Wv     = (const float*)d_in[7];
    const float* g0     = (const float*)d_in[8];
    const float* beta0  = (const float*)d_in[9];
    const float* W1     = (const float*)d_in[10];
    const float* b1     = (const float*)d_in[11];
    const float* g1     = (const float*)d_in[12];
    const float* beta1  = (const float*)d_in[13];
    const float* Wf1    = (const float*)d_in[14];
    const float* bf1    = (const float*)d_in[15];
    const float* Wf2    = (const float*)d_in[16];
    const float* bf2    = (const float*)d_in[17];

    // ---- workspace carve-out -------------------------------------------
    char* ws = (char*)d_ws;
    size_t off = 0;
    auto carve = [&](size_t bytes) -> void* {
        void* p = ws + off;
        off = (off + bytes + 255) & ~(size_t)255;
        return p;
    };
    float*     seq   = (float*)    carve((size_t)TTOK * EDIM * 4);          // 121 MB
    _Float16*  Kbuf  = (_Float16*) carve((size_t)HHEADS * TTOK * EDIM * 2); // 121 MB
    _Float16*  Vbuf  = (_Float16*) carve((size_t)HHEADS * TTOK * EDIM * 2); // 121 MB
    float*     Q0    = (float*)    carve((size_t)HHEADS * BSZ * EDIM * 4);
    float*     msa0  = (float*)    carve((size_t)BSZ * 600 * 4);
    float*     cat1  = (float*)    carve((size_t)BSZ * 900 * 4);
    float*     y1    = (float*)    carve((size_t)BSZ * 300 * 4);
    float*     cat2  = (float*)    carve((size_t)BSZ * 1200 * 4);
    float*     fc1   = (float*)    carve((size_t)BSZ * 512 * 4);
    float*     st0   = (float*)    carve(64);
    float*     st1   = (float*)    carve(64);

    // 1) cls row + positional halves of seq
    {
        const long total = (long)BSZ * NPOS * LD;
        fill_seq_kernel<<<(unsigned)((total + 255) / 256), 256, 0, stream>>>(seq, pos_tab, cls_emb);
    }

    // 2) projection: relu(x @ W_proj + b_proj) -> seq[b, 1+p, 0:150] (row-remap)
    gemm_wmma_kernel<float, true, true>
        <<<dim3(ceil_div(LD, 64), ceil_div(MPROJ, 64)), 256, 0, stream>>>(
            x, PD, W_proj, LD, seq, EDIM, b_proj, MPROJ, LD, PD, P196);

    // 3) K and V for both heads (full sequence), stored f16
    for (int h = 0; h < HHEADS; ++h) {
        gemm_wmma_kernel<_Float16, false, false>
            <<<dim3(ceil_div(EDIM, 64), ceil_div(TTOK, 64)), 256, 0, stream>>>(
                seq, EDIM, Wk + (long)h * EDIM * EDIM, EDIM,
                Kbuf + (long)h * TTOK * EDIM, EDIM, nullptr, TTOK, EDIM, EDIM, 0);
        gemm_wmma_kernel<_Float16, false, false>
            <<<dim3(ceil_div(EDIM, 64), ceil_div(TTOK, 64)), 256, 0, stream>>>(
                seq, EDIM, Wv + (long)h * EDIM * EDIM, EDIM,
                Vbuf + (long)h * TTOK * EDIM, EDIM, nullptr, TTOK, EDIM, EDIM, 0);
    }

    // 4) Q only for n = 0 (lda strides over whole sequences)
    for (int h = 0; h < HHEADS; ++h) {
        gemm_wmma_kernel<float, false, false>
            <<<dim3(ceil_div(EDIM, 64), ceil_div(BSZ, 64)), 256, 0, stream>>>(
                seq, NPOS * EDIM, Wq + (long)h * EDIM * EDIM, EDIM,
                Q0 + (long)h * BSZ * EDIM, EDIM, nullptr, BSZ, EDIM, EDIM, 0);
    }

    // 5) attention row 0 per (b, h) -> msa0 (b, 600)
    attn_row0_kernel<<<BSZ * HHEADS, 256, 0, stream>>>(Kbuf, Vbuf, Q0, msa0);

    // 6) BN0 (position-0 stats) + concat with skip1
    bn_stats_kernel<<<1, 256, 0, stream>>>(msa0, BSZ * 600, st0);
    build_cat1_kernel<<<ceil_div(BSZ * 900, 256), 256, 0, stream>>>(cat1, msa0, seq, st0, g0, beta0);

    // 7) y1 = relu(cat1 @ W1 + b1)
    gemm_wmma_kernel<float, true, false>
        <<<dim3(ceil_div(300, 64), ceil_div(BSZ, 64)), 256, 0, stream>>>(
            cat1, 900, W1, 300, y1, 300, b1, BSZ, 300, 900, 0);

    // 8) BN1 + concat with skip2
    bn_stats_kernel<<<1, 256, 0, stream>>>(y1, BSZ * 300, st1);
    build_cat2_kernel<<<ceil_div(BSZ * 1200, 256), 256, 0, stream>>>(cat2, y1, cat1, st1, g1, beta1);

    // 9) classifier head
    gemm_wmma_kernel<float, false, false>
        <<<dim3(ceil_div(512, 64), ceil_div(BSZ, 64)), 256, 0, stream>>>(
            cat2, 1200, Wf1, 512, fc1, 512, bf1, BSZ, 512, 1200, 0);
    gemm_wmma_kernel<float, false, false>
        <<<dim3(ceil_div(CDIM, 64), ceil_div(BSZ, 64)), 256, 0, stream>>>(
            fc1, 512, Wf2, CDIM, (float*)d_out, CDIM, bf2, BSZ, CDIM, 512, 0);
}
